// LinearPFN_old_45097156608183
// MI455X (gfx1250) — matrix operand order repaired and checked by reading, not compiled, not run on hardware
//
#include <hip/hip_runtime.h>
#include <cstdint>
#include <cstddef>

// ---------------- model dims (match reference) ----------------
constexpr int B_ = 2, C_ = 1536, Q_ = 512, S_ = 2048;
constexpr int D_ = 256, NH_ = 8, DH_ = 32, DFF_ = 1024, NL_ = 6;
constexpr int LIN_ = 20, HOUT_ = 10;
constexpr int BS_ = B_ * S_;                 // 4096 rows for all big GEMMs
constexpr float NEGBIG = -1e9f;
constexpr float SCALE_ = 0.17677669529663687f;  // 1/sqrt(32)

typedef __bf16 v16bf __attribute__((ext_vector_type(16)));
typedef float  v8f   __attribute__((ext_vector_type(8)));

union FragU { v16bf v; uint4 q[2]; };

#define DEV __device__ __forceinline__

DEV v8f wmma_bf16(v16bf a, v16bf b, v8f c) {
  // D(16x16,f32) = A(16x32,bf16) x B(32x16,bf16) + C
  return __builtin_amdgcn_wmma_f32_16x16x32_bf16(false, a, false, b,
                                                 (short)0, c, false, false);
}

// A fragment: 16x32 bf16, row-major source with leading dim `ld`.
// ISA layout: lane L holds row (L&15); lanes<16 K={0..7,16..23}, lanes>=16 K={8..15,24..31}
DEV v16bf load_a_bf16(const __bf16* base, int ld, int row0, int k0) {
  const int lane = threadIdx.x & 31;
  const int row  = row0 + (lane & 15);
  const int off  = (lane >> 4) * 8;
  const __bf16* p = base + (size_t)row * ld + k0;
  FragU u;
  u.q[0] = *reinterpret_cast<const uint4*>(p + off);        // K = off..off+7
  u.q[1] = *reinterpret_cast<const uint4*>(p + 16 + off);   // K = 16+off..23+off
  return u.v;
}

// B fragment: 32x16 bf16 where B[k][n] = W[col0+n][k0+k], W row-major [N,K] ld=`ld`.
// ISA layout: lane L holds col (L&15); lanes<16 K={0..7,16..23}, lanes>=16 K={8..15,24..31}
DEV v16bf load_b_bf16(const __bf16* base, int ld, int col0, int k0) {
  const int lane = threadIdx.x & 31;
  const int col  = col0 + (lane & 15);
  const int koff = (lane >> 4) * 8;
  const __bf16* p = base + (size_t)col * ld + k0 + koff;
  FragU u;
  u.q[0] = *reinterpret_cast<const uint4*>(p);
  u.q[1] = *reinterpret_cast<const uint4*>(p + 16);
  return u.v;
}

DEV float rowmax16(float v) {
  #pragma unroll
  for (int o = 1; o < 16; o <<= 1) v = fmaxf(v, __shfl_xor(v, o, 16));
  return v;
}
DEV float rowsum16(float v) {
  #pragma unroll
  for (int o = 1; o < 16; o <<= 1) v += __shfl_xor(v, o, 16);
  return v;
}

DEV int t_at(const int* tctx, const int* tqry, int b, int s) {
  return (s < C_) ? tctx[b * C_ + s] : tqry[b * Q_ + (s - C_)];
}

// Issue one 16-byte async global->LDS DMA (CDNA5 GLOBAL_LOAD_ASYNC_TO_LDS_B128,
// tracked on ASYNCcnt). ldsAddr = LDS byte address (low 32 bits of flat shared addr).
DEV void async_g2l_b128(uint32_t ldsAddr, const void* gaddr) {
  asm volatile("global_load_async_to_lds_b128 %0, %1, off"
               :: "v"(ldsAddr), "v"((uint64_t)(uintptr_t)gaddr)
               : "memory");
}
DEV void wait_async0() {
  asm volatile("s_wait_asynccnt 0" ::: "memory");
}

// ---------------- fp32 -> bf16 ----------------
__global__ void f2bf_kernel(const float* __restrict__ in, __bf16* __restrict__ out, int n) {
  int i = blockIdx.x * 256 + threadIdx.x;
  if (i < n) out[i] = (__bf16)in[i];
}

// ---------------- embedding: Z = concat(ctx_in@Wc^T+bc, qry_in@Wq^T+bq) ----------------
__global__ __launch_bounds__(256) void embed_kernel(
    const float* __restrict__ ctx_x, const float* __restrict__ ctx_z,
    const float* __restrict__ qry_x,
    const float* __restrict__ Wc, const float* __restrict__ bc,
    const float* __restrict__ Wq, const float* __restrict__ bq,
    float* __restrict__ Zf, __bf16* __restrict__ Zb) {
  const int bs = blockIdx.x;            // 0..B*S-1
  const int b = bs / S_, s = bs % S_;
  const int d = threadIdx.x;            // 0..255
  float acc;
  if (s < C_) {
    acc = bc[d];
    const float* xr = ctx_x + ((size_t)b * C_ + s) * LIN_;
    const float* zr = ctx_z + ((size_t)b * C_ + s) * HOUT_;
    const float* wr = Wc + (size_t)d * (LIN_ + HOUT_);
    #pragma unroll
    for (int k = 0; k < LIN_; k++)  acc += xr[k] * wr[k];
    #pragma unroll
    for (int k = 0; k < HOUT_; k++) acc += zr[k] * wr[LIN_ + k];
  } else {
    acc = bq[d];
    const float* xr = qry_x + ((size_t)b * Q_ + (s - C_)) * LIN_;
    const float* wr = Wq + (size_t)d * (LIN_ + HOUT_);
    #pragma unroll
    for (int k = 0; k < LIN_; k++)  acc += xr[k] * wr[k];
    // query HOUT slots are zero in the reference
  }
  const size_t o = (size_t)bs * D_ + d;
  Zf[o] = acc;
  Zb[o] = (__bf16)acc;
}

// ---------------- generic WMMA GEMM: out[M,N] = A[M,K](bf16) @ W[N,K]^T(bf16) + bias ----------------
// block = 256 threads = 8 waves; block tile 128(M) x 64(N); wave tile 32x32 (2x2 frags).
// Per K-step (32): A 128x32 and W 64x32 tiles are DMA'd global->LDS with
// GLOBAL_LOAD_ASYNC_TO_LDS_B128 (ASYNCcnt), then fragments are read from LDS
// (ds_load_b128) into v_wmma_f32_16x16x32_bf16.
__global__ __launch_bounds__(256) void gemm_bf16_kernel(
    const __bf16* __restrict__ A, const __bf16* __restrict__ W,
    const float* __restrict__ bias,
    float* __restrict__ outF, __bf16* __restrict__ outB,
    int M, int N, int K, int relu) {
  __shared__ __bf16 Atile[128][32];   // 8 KB
  __shared__ __bf16 Wtile[64][32];    // 4 KB

  const int tid  = threadIdx.x;
  const int wid  = tid >> 5;
  const int lane = tid & 31;
  const int wm = wid & 3, wn = wid >> 2;
  const int mblk = blockIdx.y * 128;
  const int nblk = blockIdx.x * 64;
  const int m0 = wm * 32;             // within block tile
  const int n0 = wn * 32;

  const uint32_t aBase = (uint32_t)(uintptr_t)&Atile[0][0];
  const uint32_t wBase = (uint32_t)(uintptr_t)&Wtile[0][0];

  v8f acc00 = {}, acc01 = {}, acc10 = {}, acc11 = {};

  for (int k0 = 0; k0 < K; k0 += 32) {
    __syncthreads();   // previous iteration's LDS reads complete before overwrite

    // A tile: 512 16B chunks, 2 per thread. chunk -> (row = ch>>2, c = ch&3)
    #pragma unroll
    for (int i = 0; i < 2; i++) {
      const int ch  = tid + i * 256;
      const int row = ch >> 2, c = ch & 3;
      async_g2l_b128(aBase + row * 64 + c * 16,
                     A + (size_t)(mblk + row) * K + k0 + c * 8);
    }
    // W tile: 256 16B chunks, 1 per thread
    {
      const int row = tid >> 2, c = tid & 3;
      async_g2l_b128(wBase + row * 64 + c * 16,
                     W + (size_t)(nblk + row) * K + k0 + c * 8);
    }
    if (k0 + 32 < K) {  // warm L2 for the next K tile (global_prefetch_b8)
      __builtin_prefetch(A + (size_t)(mblk + (tid >> 1)) * K + k0 + 32, 0, 0);
      __builtin_prefetch(W + (size_t)(nblk + (tid >> 2)) * K + k0 + 32, 0, 0);
    }

    wait_async0();     // this wave's DMAs landed in LDS
    __syncthreads();   // everyone's DMAs visible

    v16bf a0 = load_a_bf16(&Atile[0][0], 32, m0,      0);
    v16bf a1 = load_a_bf16(&Atile[0][0], 32, m0 + 16, 0);
    v16bf b0 = load_b_bf16(&Wtile[0][0], 32, n0,      0);
    v16bf b1 = load_b_bf16(&Wtile[0][0], 32, n0 + 16, 0);
    acc00 = wmma_bf16(a0, b0, acc00);
    acc01 = wmma_bf16(a0, b1, acc01);
    acc10 = wmma_bf16(a1, b0, acc10);
    acc11 = wmma_bf16(a1, b1, acc11);
  }

  const int hi = lane >> 4, nl = lane & 15;
  v8f* accs[4] = { &acc00, &acc01, &acc10, &acc11 };
  #pragma unroll
  for (int fm = 0; fm < 2; fm++) {
    #pragma unroll
    for (int fn = 0; fn < 2; fn++) {
      const v8f& cc = *accs[fm * 2 + fn];
      const int n = nblk + n0 + fn * 16 + nl;
      const float bv = bias ? bias[n] : 0.0f;
      #pragma unroll
      for (int r = 0; r < 8; r++) {
        const int m = mblk + m0 + fm * 16 + r + 8 * hi;
        float v = cc[r] + bv;
        if (relu) v = fmaxf(v, 0.0f);
        const size_t o = (size_t)m * N + n;
        if (outF) outF[o] = v;
        if (outB) outB[o] = (__bf16)v;
      }
    }
  }
}

// ---------------- qkv split/pack: fp32 [B,S,3D] -> bf16 Q/K [B,h,S,32], V^T [B,h,32,S] ----------------
__global__ void qkv_pack_kernel(const float* __restrict__ qkv,
                                __bf16* __restrict__ Qb, __bf16* __restrict__ Kb,
                                __bf16* __restrict__ Vt) {
  const int idx = blockIdx.x * 256 + threadIdx.x;
  if (idx >= BS_ * D_) return;
  const int d = idx % D_;
  const int s = (idx / D_) % S_;
  const int b = idx / (D_ * S_);
  const int h = d / DH_, dh = d % DH_;
  const float* base = qkv + ((size_t)b * S_ + s) * (3 * D_);
  const size_t qko = (((size_t)b * NH_ + h) * S_ + s) * DH_ + dh;
  Qb[qko] = (__bf16)base[d];
  Kb[qko] = (__bf16)base[D_ + d];
  Vt[(((size_t)b * NH_ + h) * DH_ + dh) * S_ + s] = (__bf16)base[2 * D_ + d];
}

// ---------------- flash attention: one wave per 16-row tile ----------------
__global__ __launch_bounds__(256) void attn_kernel(
    const __bf16* __restrict__ Qh, const __bf16* __restrict__ Kh,
    const __bf16* __restrict__ Vt,
    const int* __restrict__ tctx, const int* __restrict__ tqry,
    const float* __restrict__ alpha, __bf16* __restrict__ outBf) {
  __shared__ __bf16 pbuf[8][16][32];   // per-wave P tile (bf16)

  const int tid = threadIdx.x, wid = tid >> 5, lane = tid & 31;
  const int tile = blockIdx.x * 8 + wid;          // 0 .. B*NH*(S/16)-1
  const int b  = tile / (NH_ * (S_ / 16));
  const int rm = tile % (NH_ * (S_ / 16));
  const int h  = rm / (S_ / 16);
  const int s0 = (rm % (S_ / 16)) * 16;

  const __bf16* Qp = Qh + ((size_t)b * NH_ + h) * S_ * DH_;
  const __bf16* Kp = Kh + ((size_t)b * NH_ + h) * S_ * DH_;
  const __bf16* Vp = Vt + ((size_t)b * NH_ + h) * DH_ * S_;
  const float alph = alpha[h];
  const int hi = lane >> 4, nl = lane & 15;

  int ti[8];
  #pragma unroll
  for (int r = 0; r < 8; r++) ti[r] = t_at(tctx, tqry, b, s0 + r + 8 * hi);

  const v16bf qa = load_a_bf16(Qp, DH_, s0, 0);

  v8f o0 = {}, o1 = {};
  float rmax[8], rsum[8];
  #pragma unroll
  for (int r = 0; r < 8; r++) { rmax[r] = -1e30f; rsum[r] = 0.0f; }

  for (int j0 = 0; j0 < S_; j0 += 32) {
    // scores: 16 rows x 32 cols via two WMMAs (K-reduction = DH = 32)
    v16bf kb0 = load_b_bf16(Kp, DH_, j0,      0);
    v16bf kb1 = load_b_bf16(Kp, DH_, j0 + 16, 0);
    v8f z = {};
    v8f sc0 = wmma_bf16(qa, kb0, z);
    v8f sc1 = wmma_bf16(qa, kb1, z);

    const int jc0 = j0 + nl, jc1 = jc0 + 16;
    const int tj0 = t_at(tctx, tqry, b, jc0);
    const int tj1 = t_at(tctx, tqry, b, jc1);
    const bool st0 = (jc0 >= C_), st1 = (jc1 >= C_);

    float sv0[8], sv1[8];
    #pragma unroll
    for (int r = 0; r < 8; r++) {
      float a0 = sc0[r] * SCALE_ - alph * fabsf((float)(ti[r] - tj0));
      float a1 = sc1[r] * SCALE_ - alph * fabsf((float)(ti[r] - tj1));
      if (st0 || ti[r] < tj0) a0 = NEGBIG;
      if (st1 || ti[r] < tj1) a1 = NEGBIG;
      sv0[r] = a0; sv1[r] = a1;
    }

    // online softmax update
    float fac[8], mnew[8];
    #pragma unroll
    for (int r = 0; r < 8; r++) {
      float lm = rowmax16(fmaxf(sv0[r], sv1[r]));
      mnew[r] = fmaxf(rmax[r], lm);
      fac[r] = __expf(rmax[r] - mnew[r]);
      rmax[r] = mnew[r];
      o0[r] = o0[r] * fac[r];
      o1[r] = o1[r] * fac[r];
    }
    #pragma unroll
    for (int r = 0; r < 8; r++) {
      const float p0 = __expf(sv0[r] - mnew[r]);
      const float p1 = __expf(sv1[r] - mnew[r]);
      const int m = r + 8 * hi;
      pbuf[wid][m][nl]      = (__bf16)p0;
      pbuf[wid][m][16 + nl] = (__bf16)p1;
      rsum[r] = rsum[r] * fac[r] + rowsum16(p0 + p1);
    }

    // drain our LDS stores, reload P in A-fragment layout (ds_load_b128)
    asm volatile("s_wait_dscnt 0" ::: "memory");
    v16bf pa  = load_a_bf16(&pbuf[wid][0][0], 32, 0, 0);
    v16bf vb0 = load_b_bf16(Vp, S_, 0,  j0);   // V^T: B[k=j][n=d] contiguous in j
    v16bf vb1 = load_b_bf16(Vp, S_, 16, j0);
    o0 = wmma_bf16(pa, vb0, o0);
    o1 = wmma_bf16(pa, vb1, o1);
  }

  // normalize and write bf16 [B,S,D] at column block h*32
  #pragma unroll
  for (int r = 0; r < 8; r++) {
    const float inv = 1.0f / rsum[r];
    const int m = r + 8 * hi;
    const size_t ro = ((size_t)b * S_ + s0 + m) * D_ + h * DH_;
    outBf[ro + nl]      = (__bf16)(o0[r] * inv);
    outBf[ro + 16 + nl] = (__bf16)(o1[r] * inv);
  }
}

// ---------------- residual add + layernorm (one row per block) ----------------
__global__ __launch_bounds__(256) void add_ln_kernel(
    float* __restrict__ Z, const float* __restrict__ addv,
    const float* __restrict__ sca, const float* __restrict__ beta,
    __bf16* __restrict__ Zb) {
  __shared__ float red[256];
  const int row = blockIdx.x, tid = threadIdx.x;
  const size_t o = (size_t)row * D_ + tid;
  const float x = Z[o] + addv[o];
  red[tid] = x;
  __syncthreads();
  #pragma unroll
  for (int s = 128; s > 0; s >>= 1) {
    if (tid < s) red[tid] += red[tid + s];
    __syncthreads();
  }
  const float mu = red[0] * (1.0f / D_);
  __syncthreads();
  const float d = x - mu;
  red[tid] = d * d;
  __syncthreads();
  #pragma unroll
  for (int s = 128; s > 0; s >>= 1) {
    if (tid < s) red[tid] += red[tid + s];
    __syncthreads();
  }
  const float var = red[0] * (1.0f / D_);
  const float y = d * rsqrtf(var + 1e-5f) * sca[tid] + beta[tid];
  Z[o] = y;
  Zb[o] = (__bf16)y;
}

// ---------------- final head: out[b,q,h] = Z[b,C+q,:] . W_out[h,:] + b_out[h] ----------------
__global__ void head_out_kernel(const float* __restrict__ Z,
                                const float* __restrict__ Wo,
                                const float* __restrict__ bo,
                                float* __restrict__ out) {
  const int idx = blockIdx.x * 256 + threadIdx.x;
  if (idx >= B_ * Q_ * HOUT_) return;
  const int h = idx % HOUT_;
  const int q = (idx / HOUT_) % Q_;
  const int b = idx / (HOUT_ * Q_);
  const float* zr = Z + ((size_t)b * S_ + C_ + q) * D_;
  const float* wr = Wo + (size_t)h * D_;
  float acc = bo[h];
  #pragma unroll 8
  for (int d = 0; d < D_; d++) acc += zr[d] * wr[d];
  out[idx] = acc;
}

// ---------------- host launcher ----------------
extern "C" void kernel_launch(void* const* d_in, const int* in_sizes, int n_in,
                              void* d_out, int out_size, void* d_ws, size_t ws_size,
                              hipStream_t stream) {
  (void)in_sizes; (void)n_in; (void)out_size; (void)ws_size;

  const float* ctx_x = (const float*)d_in[0];
  const float* ctx_z = (const float*)d_in[1];
  const float* qry_x = (const float*)d_in[2];
  const int*   t_ctx = (const int*)d_in[3];
  const int*   t_qry = (const int*)d_in[4];
  const float* W_ctx = (const float*)d_in[5];
  const float* b_ctx = (const float*)d_in[6];
  const float* W_qry = (const float*)d_in[7];
  const float* b_qry = (const float*)d_in[8];
  const float* alpha = (const float*)d_in[9];
  const float* Wqkv  = (const float*)d_in[10];
  const float* bqkv  = (const float*)d_in[11];
  const float* Wo    = (const float*)d_in[12];
  const float* bo    = (const float*)d_in[13];
  const float* ln1s  = (const float*)d_in[14];
  const float* ln1b  = (const float*)d_in[15];
  const float* W1    = (const float*)d_in[16];
  const float* b1    = (const float*)d_in[17];
  const float* W2    = (const float*)d_in[18];
  const float* b2    = (const float*)d_in[19];
  const float* ln2s  = (const float*)d_in[20];
  const float* ln2b  = (const float*)d_in[21];
  const float* W_out = (const float*)d_in[22];
  const float* b_out = (const float*)d_in[23];

  size_t off = 0;
  auto alloc = [&](size_t bytes) -> void* {
    off = (off + 255) & ~(size_t)255;
    void* p = (char*)d_ws + off;
    off += bytes;
    return p;
  };

  float*  Zf    = (float*)alloc((size_t)BS_ * D_ * 4);
  __bf16* Zb    = (__bf16*)alloc((size_t)BS_ * D_ * 2);
  float*  qkvF  = (float*)alloc((size_t)BS_ * 3 * D_ * 4);
  __bf16* Qb    = (__bf16*)alloc((size_t)BS_ * D_ * 2);
  __bf16* Kb    = (__bf16*)alloc((size_t)BS_ * D_ * 2);
  __bf16* Vtb   = (__bf16*)alloc((size_t)BS_ * D_ * 2);
  __bf16* attnB = (__bf16*)alloc((size_t)BS_ * D_ * 2);
  float*  tmpF  = (float*)alloc((size_t)BS_ * D_ * 4);
  __bf16* Hb    = (__bf16*)alloc((size_t)BS_ * DFF_ * 2);
  __bf16* WqkvB = (__bf16*)alloc((size_t)NL_ * 3 * D_ * D_ * 2);
  __bf16* WoB   = (__bf16*)alloc((size_t)NL_ * D_ * D_ * 2);
  __bf16* W1B   = (__bf16*)alloc((size_t)NL_ * DFF_ * D_ * 2);
  __bf16* W2B   = (__bf16*)alloc((size_t)NL_ * D_ * DFF_ * 2);

  auto cvt = [&](const float* src, __bf16* dst, int n) {
    f2bf_kernel<<<(n + 255) / 256, 256, 0, stream>>>(src, dst, n);
  };
  cvt(Wqkv, WqkvB, NL_ * 3 * D_ * D_);
  cvt(Wo,   WoB,   NL_ * D_ * D_);
  cvt(W1,   W1B,   NL_ * DFF_ * D_);
  cvt(W2,   W2B,   NL_ * D_ * DFF_);

  embed_kernel<<<B_ * S_, D_, 0, stream>>>(ctx_x, ctx_z, qry_x,
                                           W_ctx, b_ctx, W_qry, b_qry, Zf, Zb);

  for (int l = 0; l < NL_; l++) {
    // qkv = Z @ Wqkv[l]^T + bqkv[l]
    gemm_bf16_kernel<<<dim3(3 * D_ / 64, BS_ / 128), 256, 0, stream>>>(
        Zb, WqkvB + (size_t)l * 3 * D_ * D_, bqkv + (size_t)l * 3 * D_,
        qkvF, nullptr, BS_, 3 * D_, D_, 0);

    qkv_pack_kernel<<<(BS_ * D_ + 255) / 256, 256, 0, stream>>>(qkvF, Qb, Kb, Vtb);

    attn_kernel<<<(B_ * NH_ * (S_ / 16)) / 8, 256, 0, stream>>>(
        Qb, Kb, Vtb, t_ctx, t_qry, alpha, attnB);

    // attn projection
    gemm_bf16_kernel<<<dim3(D_ / 64, BS_ / 128), 256, 0, stream>>>(
        attnB, WoB + (size_t)l * D_ * D_, bo + (size_t)l * D_,
        tmpF, nullptr, BS_, D_, D_, 0);

    add_ln_kernel<<<BS_, D_, 0, stream>>>(Zf, tmpF,
        ln1s + (size_t)l * D_, ln1b + (size_t)l * D_, Zb);

    // FFN
    gemm_bf16_kernel<<<dim3(DFF_ / 64, BS_ / 128), 256, 0, stream>>>(
        Zb, W1B + (size_t)l * DFF_ * D_, b1 + (size_t)l * DFF_,
        nullptr, Hb, BS_, DFF_, D_, 1);

    gemm_bf16_kernel<<<dim3(D_ / 64, BS_ / 128), 256, 0, stream>>>(
        Hb, W2B + (size_t)l * D_ * DFF_, b2 + (size_t)l * D_,
        tmpF, nullptr, BS_, D_, DFF_, 0);

    add_ln_kernel<<<BS_, D_, 0, stream>>>(Zf, tmpF,
        ln2s + (size_t)l * D_, ln2b + (size_t)l * D_, Zb);
  }

  head_out_kernel<<<(B_ * Q_ * HOUT_ + 255) / 256, 256, 0, stream>>>(
      Zf, W_out, b_out, (float*)d_out);
}
